// Dense_BinaryLayer_18897856102730
// MI455X (gfx1250) — matrix operand order repaired and checked by reading, not compiled, not run on hardware
//
#include <hip/hip_runtime.h>

// out = x @ sign(W) + b  — binarized-weight dense layer, f16 WMMA w/ f32 accumulate.
// x:[R,K] f32, W:[K,NC] f32 (binarized to +/-1 on the fly), b:[NC], out:[R,NC] f32.

typedef __attribute__((ext_vector_type(16))) _Float16 v16h;
typedef __attribute__((ext_vector_type(8)))  _Float16 v8h;
typedef __attribute__((ext_vector_type(4)))  _Float16 v4h;
typedef __attribute__((ext_vector_type(8)))  float    v8f;

#define TM 128   // block tile M
#define TN 128   // block tile N
#define TK 32    // K step = one v_wmma_f32_16x16x32_f16 per 16x16 subtile
#define LSTR 40  // LDS row stride in halves (32 + 8 pad = 80B, 16B-aligned rows)

__global__ __launch_bounds__(256)
void binary_dense_wmma(const float* __restrict__ X,
                       const float* __restrict__ W,
                       const float* __restrict__ BIAS,
                       float* __restrict__ OUT,
                       int R, int K, int NC)
{
    // double-buffered tiles: A = x (f16), B = sign(W) stored transposed [n][k]
    __shared__ _Float16 Alds[2][TM * LSTR];
    __shared__ _Float16 Blds[2][TN * LSTR];

    const int tid   = threadIdx.x;
    const int lane  = tid & 31;
    const int wave  = tid >> 5;
    const int wm    = wave >> 2;   // 0..1 -> 64-row slab
    const int wn    = wave & 3;    // 0..3 -> 32-col slab
    const int lmod  = lane & 15;
    const int lhalf = lane >> 4;

    const int m0 = blockIdx.y * TM;
    const int n0 = blockIdx.x * TN;

    // per-thread staging coordinates (256 threads move a 128x32 + 32x128 f32 tile)
    const int xc = tid & 7;    // float4 column within X tile row
    const int xr = tid >> 3;   // X tile row base (0..31), +32*p
    const int wc = tid & 31;   // float4 column within W tile row
    const int wr = tid >> 5;   // W tile row base (0..7), +8*p

    float4 xs[4], ws[4];

    auto load_tiles = [&](int k0) {
        #pragma unroll
        for (int p = 0; p < 4; ++p) {
            xs[p] = *(const float4*)(X + (size_t)(m0 + xr + 32 * p) * K  + (k0 + xc * 4));
            ws[p] = *(const float4*)(W + (size_t)(k0 + wr + 8  * p) * NC + (n0 + wc * 4));
        }
    };
    auto store_tiles = [&](int buf) {
        #pragma unroll
        for (int p = 0; p < 4; ++p) {
            // x tile: convert f32 -> f16, row-major [m][k]
            v4h hx = { (_Float16)xs[p].x, (_Float16)xs[p].y,
                       (_Float16)xs[p].z, (_Float16)xs[p].w };
            *(v4h*)&Alds[buf][(xr + 32 * p) * LSTR + xc * 4] = hx;
            // W tile: binarize (w>0 -> +1, else -1; matches round-half-even at 0.5),
            // store transposed [n][k] so B fragments are contiguous in k
            const int kr = wr + 8 * p;
            _Float16* bp = &Blds[buf][0];
            bp[(wc * 4 + 0) * LSTR + kr] = (ws[p].x > 0.f) ? (_Float16)1.f : (_Float16)-1.f;
            bp[(wc * 4 + 1) * LSTR + kr] = (ws[p].y > 0.f) ? (_Float16)1.f : (_Float16)-1.f;
            bp[(wc * 4 + 2) * LSTR + kr] = (ws[p].z > 0.f) ? (_Float16)1.f : (_Float16)-1.f;
            bp[(wc * 4 + 3) * LSTR + kr] = (ws[p].w > 0.f) ? (_Float16)1.f : (_Float16)-1.f;
        }
    };

    v8f acc[4][2];
    #pragma unroll
    for (int i = 0; i < 4; ++i)
        #pragma unroll
        for (int j = 0; j < 2; ++j)
            #pragma unroll
            for (int e = 0; e < 8; ++e) acc[i][j][e] = 0.0f;

    const int nk = K / TK;
    load_tiles(0);
    store_tiles(0);
    __syncthreads();

    for (int kt = 0; kt < nk; ++kt) {
        const int cur = kt & 1;
        if (kt + 1 < nk) {
            load_tiles((kt + 1) * TK);                 // overlap with compute below
            if (kt + 2 < nk) {                          // gfx1250 global_prefetch_b8
                __builtin_prefetch(X + (size_t)(m0 + xr) * K  + ((kt + 2) * TK + xc * 4), 0, 3);
                __builtin_prefetch(W + (size_t)((kt + 2) * TK + wr) * NC + (n0 + wc * 4), 0, 3);
            }
        }

        // Gather fragments per the documented 16-bit WMMA VGPR layouts.
        union U16 { v16h v; v8h h[2]; };
        v16h afr[4], bfr[2];
        #pragma unroll
        for (int mt = 0; mt < 4; ++mt) {
            // A 16x32: lanes 0-15 -> k 0..7 & 16..23 ; lanes 16-31 -> k 8..15 & 24..31
            const _Float16* ap = &Alds[cur][(wm * 64 + mt * 16 + lmod) * LSTR + lhalf * 8];
            U16 u; u.h[0] = *(const v8h*)ap; u.h[1] = *(const v8h*)(ap + 16);
            afr[mt] = u.v;
        }
        #pragma unroll
        for (int nt = 0; nt < 2; ++nt) {
            // B 32x16: lane = n, lanes 0-15 -> k 0..7 & 16..23 ; lanes 16-31 -> k 8..15 & 24..31
            const _Float16* bp = &Blds[cur][(wn * 32 + nt * 16 + lmod) * LSTR + lhalf * 8];
            U16 u; u.h[0] = *(const v8h*)bp; u.h[1] = *(const v8h*)(bp + 16);
            bfr[nt] = u.v;
        }

        #pragma unroll
        for (int mt = 0; mt < 4; ++mt)
            #pragma unroll
            for (int nt = 0; nt < 2; ++nt)
                acc[mt][nt] = __builtin_amdgcn_wmma_f32_16x16x32_f16(
                    false, afr[mt], false, bfr[nt],
                    (short)0, acc[mt][nt], false, false);

        if (kt + 1 < nk) store_tiles(cur ^ 1);  // buf^1 fully consumed last iter
        __syncthreads();
    }

    // Epilogue: C/D layout -> lane = n (0..15), VGPR r = m (lanes 16-31: m+8). Add bias.
    #pragma unroll
    for (int nt = 0; nt < 2; ++nt) {
        const int n  = n0 + wn * 32 + nt * 16 + lmod;
        const float bv = BIAS[n];
        #pragma unroll
        for (int mt = 0; mt < 4; ++mt) {
            const int mbase = m0 + wm * 64 + mt * 16 + lhalf * 8;
            #pragma unroll
            for (int r = 0; r < 8; ++r)
                OUT[(size_t)(mbase + r) * NC + n] = acc[mt][nt][r] + bv;
        }
    }
}

extern "C" void kernel_launch(void* const* d_in, const int* in_sizes, int n_in,
                              void* d_out, int out_size, void* d_ws, size_t ws_size,
                              hipStream_t stream) {
    const float* x = (const float*)d_in[0];
    const float* W = (const float*)d_in[1];
    const float* b = (const float*)d_in[2];
    float* out     = (float*)d_out;

    const int NC = in_sizes[2];            // d_out = 1024 (bias length)
    const int K  = in_sizes[1] / NC;       // d_in  = 1024
    const int R  = in_sizes[0] / K;        // N     = 16384

    dim3 grid(NC / TN, R / TM);            // 8 x 128 blocks
    binary_dense_wmma<<<grid, 256, 0, stream>>>(x, W, b, out, R, K, NC);
}
